// GATConvwithEdgeFeat_69415261438028
// MI455X (gfx1250) — matrix operand order repaired and checked by reading, not compiled, not run on hardware
//
#include <hip/hip_runtime.h>
#include <hip/hip_bf16.h>

// ---------------------------------------------------------------------------
// EdgeGATConv forward for MI455X (gfx1250, wave32, WMMA).
//
// Roofline: edge GEMM = 105 GFLOP; min HBM traffic ~1.5-2 GB => ~80 us at
// 23.3 TB/s. bf16 WMMA (16x16x32, f32 accum) keeps the GEMM at the memory
// roofline; fp32 WMMA (16x16x4) would be ~8x slower per instruction and
// compute-bound.
//
// This revision:
//  * Weights pre-converted to bf16 once (128 KB each), then staged into LDS
//    per workgroup with GLOBAL_LOAD_ASYNC_TO_LDS_B128 (ASYNCcnt-tracked,
//    per-lane LDS addresses insert the 8-bf16 row padding during the copy).
//  * Multiple 16-row tiles per wave (4 edge / 2 node) to amortize the LDS
//    weight staging: edge staging traffic drops 6250->1563 WG * 128 KB.
//  * Attention logits el/er/ee fused into the GEMM epilogue via wave32
//    __shfl_xor butterfly reductions; feat_e stored bf16 to halve re-reads.
//  * Segment softmax: monotonic-uint atomicMax, atomicAdd denominators,
//    per-edge scatter-add into d_out (51 MB dst accumulator, L2-resident).
// ---------------------------------------------------------------------------

typedef __attribute__((ext_vector_type(16))) __bf16 bf16x16;
typedef __attribute__((ext_vector_type(8)))  __bf16 bf16x8;
typedef __attribute__((ext_vector_type(8)))  float  f32x8;

#define IN_FEAT     256
#define LDS_PITCH   264                       // 256 + 8 bf16 pad per row
#define SHMEM_BYTES (256 * LDS_PITCH * 2)     // 135168 B < 320KB WGP LDS
#define ETPW        4                         // edge tiles per wave
#define NTPW        2                         // node tiles per wave

// ---------------- float <-> order-preserving uint (for atomic max) ---------
__device__ __forceinline__ unsigned enc_f32(float f) {
  unsigned u = __float_as_uint(f);
  return (u & 0x80000000u) ? ~u : (u | 0x80000000u);
}
__device__ __forceinline__ float dec_f32(unsigned k) {
  return (k & 0x80000000u) ? __uint_as_float(k & 0x7fffffffu)
                           : __uint_as_float(~k);
}

// ---------------- async LDS staging of pre-converted bf16 weights ----------
// Copies 256x256 bf16 (128 KB) into LDS with LDS_PITCH row padding using
// gfx1250 async global->LDS DMA (GVS mode: saddr base + 32-bit vaddr offset).
// Dynamic LDS base is 0 here (no static LDS), so raw byte offsets are valid
// LDS addresses. 32 async instructions per wave (< 63 ASYNCcnt limit).
__device__ __forceinline__ void stage_weights_async(const __bf16* wbf) {
#pragma unroll
  for (int it = 0; it < 32; ++it) {
    int chunk = threadIdx.x + (it << 8);                 // 0..8191 16B chunks
    int row = chunk >> 5;                                // 32 chunks per row
    int c   = chunk & 31;
    unsigned lds_off = (unsigned)(row * (LDS_PITCH * 2) + (c << 4));
    unsigned g_off   = (unsigned)(chunk << 4);
    asm volatile("global_load_async_to_lds_b128 %0, %1, %2"
                 :: "v"(lds_off), "v"(g_off), "s"(wbf)
                 : "memory");
  }
#if __has_builtin(__builtin_amdgcn_s_wait_asynccnt)
  __builtin_amdgcn_s_wait_asynccnt(0);
#else
  asm volatile("s_wait_asynccnt 0x0" ::: "memory");
#endif
  __syncthreads();
}

// ---------------- load A fragment: one 16x32 K-step of a 16-row tile -------
// Lane layout (ISA 7.12.2, 16-bit A 16x32): lane<16 holds K in
// [ks*32+0,8)U[ks*32+16,24); lane>=16 holds [+8,+16)U[+24,+32).
__device__ __forceinline__ bf16x16 load_a_frag(const float4* arow, int ks, int sel) {
  int b4 = ks * 8 + sel * 2;             // float4 index of first 8-chunk
  float4 c0 = arow[b4];
  float4 c1 = arow[b4 + 1];
  float4 c2 = arow[b4 + 4];              // +16 floats
  float4 c3 = arow[b4 + 5];
  bf16x16 t;
  t[0]  = (__bf16)c0.x; t[1]  = (__bf16)c0.y; t[2]  = (__bf16)c0.z; t[3]  = (__bf16)c0.w;
  t[4]  = (__bf16)c1.x; t[5]  = (__bf16)c1.y; t[6]  = (__bf16)c1.z; t[7]  = (__bf16)c1.w;
  t[8]  = (__bf16)c2.x; t[9]  = (__bf16)c2.y; t[10] = (__bf16)c2.z; t[11] = (__bf16)c2.w;
  t[12] = (__bf16)c3.x; t[13] = (__bf16)c3.y; t[14] = (__bf16)c3.z; t[15] = (__bf16)c3.w;
  return t;
}

// ---------------- load B fragment from LDS ---------------------------------
// Lane = output column n (= weight row nt*16+laneN), K values
// [ks*32+sel*8,+8) U [ks*32+16+sel*8,+8) (matches A layout) -> 2x ds_load_b128.
__device__ __forceinline__ bf16x16 load_b_frag(const __bf16* lds, int nt, int ks,
                                               int laneN, int sel) {
  int brow = (nt << 4) + laneN;
  const bf16x8* bp = (const bf16x8*)&lds[brow * LDS_PITCH + (ks << 5) + (sel << 3)];
  bf16x8 blo = bp[0];
  bf16x8 bhi = bp[2];
  bf16x16 b;
#pragma unroll
  for (int i = 0; i < 8; ++i) { b[i] = blo[i]; b[8 + i] = bhi[i]; }
  return b;
}

// ===========================================================================
// Weight pre-conversion: f32 -> bf16 (run once, 2x 64K elements)
// ===========================================================================
__global__ void cvt_weights_kernel(const float* __restrict__ W,
                                   const float* __restrict__ We,
                                   __bf16* __restrict__ wbf,
                                   __bf16* __restrict__ webf, int n) {
  int i = blockIdx.x * blockDim.x + threadIdx.x;
  if (i < n) { wbf[i] = (__bf16)W[i]; webf[i] = (__bf16)We[i]; }
}

// ===========================================================================
// Node projection: feat_src = node_feat @ W^T  (f32 out) + fused el, er
// ===========================================================================
__global__ void __launch_bounds__(256)
node_proj_kernel(const float* __restrict__ nf, const __bf16* __restrict__ wbf,
                 const float* __restrict__ attn_l, const float* __restrict__ attn_r,
                 float* __restrict__ feat_src, float* __restrict__ el,
                 float* __restrict__ er, int nTiles) {
  extern __shared__ __bf16 lds[];
  stage_weights_async(wbf);

  int wv = threadIdx.x >> 5;
  int lane = threadIdx.x & 31;
  int laneN = lane & 15;
  int sel = lane >> 4;
  int selRow = sel << 3;
  int waveJob = blockIdx.x * 8 + wv;

#pragma unroll 1
  for (int t = 0; t < NTPW; ++t) {
    int tile = waveJob * NTPW + t;
    if (tile >= nTiles) break;           // wave-uniform: EXEC all-1s at WMMA

    int base = tile << 4;
    const float4* arow = (const float4*)(nf + (size_t)(base + laneN) * IN_FEAT);

    bf16x16 a[8];
#pragma unroll
    for (int ks = 0; ks < 8; ++ks) a[ks] = load_a_frag(arow, ks, sel);

#pragma unroll 1
    for (int hh = 0; hh < 8; ++hh) {     // head = output cols [hh*32, hh*32+32)
      float epl[8], epr[8];
#pragma unroll
      for (int r = 0; r < 8; ++r) { epl[r] = 0.f; epr[r] = 0.f; }

#pragma unroll
      for (int half = 0; half < 2; ++half) {
        int nt = (hh << 1) + half;
        f32x8 acc = {};
#pragma unroll
        for (int ks = 0; ks < 8; ++ks) {
          bf16x16 b = load_b_frag(lds, nt, ks, laneN, sel);
          acc = __builtin_amdgcn_wmma_f32_16x16x32_bf16(
              false, a[ks], false, b, (short)0, acc, false, false);
        }
        float al = attn_l[(hh << 5) + (half << 4) + laneN];
        float ar = attn_r[(hh << 5) + (half << 4) + laneN];
        int colBase = (nt << 4) + laneN;
        float rl[8], rr[8];
#pragma unroll
        for (int r = 0; r < 8; ++r) {
          float v = acc[r];
          feat_src[(size_t)(base + selRow + r) * IN_FEAT + colBase] = v;
          rl[r] = v * al;
          rr[r] = v * ar;
        }
#pragma unroll
        for (int m = 1; m < 16; m <<= 1) {  // reduce within 16-lane half
#pragma unroll
          for (int r = 0; r < 8; ++r) {
            rl[r] += __shfl_xor(rl[r], m, 32);
            rr[r] += __shfl_xor(rr[r], m, 32);
          }
        }
#pragma unroll
        for (int r = 0; r < 8; ++r) { epl[r] += rl[r]; epr[r] += rr[r]; }
      }
      if (laneN == 0) {
#pragma unroll
        for (int r = 0; r < 8; ++r) {
          el[(size_t)(base + selRow + r) * 8 + hh] = epl[r];
          er[(size_t)(base + selRow + r) * 8 + hh] = epr[r];
        }
      }
    }
  }
}

// ===========================================================================
// Edge projection: feat_e = edge_feat @ We^T  (bf16 out) + fused ee
// ===========================================================================
__global__ void __launch_bounds__(256)
edge_proj_kernel(const float* __restrict__ ef, const __bf16* __restrict__ wbf,
                 const float* __restrict__ attn_e, __bf16* __restrict__ feat_e,
                 float* __restrict__ ee, int nTiles) {
  extern __shared__ __bf16 lds[];
  stage_weights_async(wbf);

  int wv = threadIdx.x >> 5;
  int lane = threadIdx.x & 31;
  int laneN = lane & 15;
  int sel = lane >> 4;
  int selRow = sel << 3;
  int waveJob = blockIdx.x * 8 + wv;

#pragma unroll 1
  for (int t = 0; t < ETPW; ++t) {
    int tile = waveJob * ETPW + t;
    if (tile >= nTiles) break;

    int base = tile << 4;
    const float4* arow = (const float4*)(ef + (size_t)(base + laneN) * IN_FEAT);

    bf16x16 a[8];
#pragma unroll
    for (int ks = 0; ks < 8; ++ks) a[ks] = load_a_frag(arow, ks, sel);

#pragma unroll 1
    for (int hh = 0; hh < 8; ++hh) {
      float ep[8];
#pragma unroll
      for (int r = 0; r < 8; ++r) ep[r] = 0.f;

#pragma unroll
      for (int half = 0; half < 2; ++half) {
        int nt = (hh << 1) + half;
        f32x8 acc = {};
#pragma unroll
        for (int ks = 0; ks < 8; ++ks) {
          bf16x16 b = load_b_frag(lds, nt, ks, laneN, sel);
          acc = __builtin_amdgcn_wmma_f32_16x16x32_bf16(
              false, a[ks], false, b, (short)0, acc, false, false);
        }
        float ae = attn_e[(hh << 5) + (half << 4) + laneN];
        int colBase = (nt << 4) + laneN;
        float rd[8];
#pragma unroll
        for (int r = 0; r < 8; ++r) {
          float v = acc[r];
          feat_e[(size_t)(base + selRow + r) * IN_FEAT + colBase] = (__bf16)v;
          rd[r] = v * ae;
        }
#pragma unroll
        for (int m = 1; m < 16; m <<= 1) {
#pragma unroll
          for (int r = 0; r < 8; ++r) rd[r] += __shfl_xor(rd[r], m, 32);
        }
#pragma unroll
        for (int r = 0; r < 8; ++r) ep[r] += rd[r];
      }
      if (laneN == 0) {
#pragma unroll
        for (int r = 0; r < 8; ++r)
          ee[(size_t)(base + selRow + r) * 8 + hh] = ep[r];
      }
    }
  }
}

// ===========================================================================
// Elementwise / scatter kernels
// ===========================================================================
__global__ void init_kernel(float* __restrict__ out, unsigned* __restrict__ smax,
                            float* __restrict__ denom, int nOut, int nSm) {
  int i = blockIdx.x * blockDim.x + threadIdx.x;
  if (i < nOut) out[i] = 0.f;
  if (i < nSm) { smax[i] = 0u; denom[i] = 0.f; }
}

__global__ void score_kernel(const int* __restrict__ src, const int* __restrict__ dst,
                             const float* __restrict__ el, const float* __restrict__ er,
                             const float* __restrict__ ee, float* __restrict__ sc,
                             unsigned* __restrict__ smax, int E) {
  int i = blockIdx.x * blockDim.x + threadIdx.x;
  if (i >= E * 8) return;
  int e = i >> 3, h = i & 7;
  int s = src[e], d = dst[e];
  float v = el[(size_t)s * 8 + h] + er[(size_t)d * 8 + h] + ee[i];
  v = v > 0.f ? v : 0.2f * v;            // leaky_relu, slope 0.2
  sc[i] = v;
  atomicMax(&smax[(size_t)d * 8 + h], enc_f32(v));
}

__global__ void expsum_kernel(const int* __restrict__ dst,
                              const unsigned* __restrict__ smax,
                              float* __restrict__ sc, float* __restrict__ denom,
                              int E) {
  int i = blockIdx.x * blockDim.x + threadIdx.x;
  if (i >= E * 8) return;
  int e = i >> 3, h = i & 7;
  int d = dst[e];
  float m = dec_f32(smax[(size_t)d * 8 + h]);   // key always written (d has edges)
  float x = __expf(sc[i] - m);
  sc[i] = x;
  atomicAdd(&denom[(size_t)d * 8 + h], x);
}

// one block (256 threads) per edge: channel o = threadIdx.x, head = o/32
__global__ void message_kernel(const int* __restrict__ src, const int* __restrict__ dst,
                               const float* __restrict__ sc, const float* __restrict__ denom,
                               const float* __restrict__ feat_src,
                               const __bf16* __restrict__ feat_e,
                               float* __restrict__ out, int E) {
  int e = blockIdx.x;
  if (e >= E) return;
  int o = threadIdx.x;
  int h = o >> 5;
  int s = src[e], d = dst[e];
  float alpha = sc[(size_t)e * 8 + h] / denom[(size_t)d * 8 + h];
  float v = (feat_src[(size_t)s * IN_FEAT + o] +
             (float)feat_e[(size_t)e * IN_FEAT + o]) * alpha;
  atomicAdd(&out[(size_t)d * IN_FEAT + o], v);
}

__global__ void finalize_kernel(const float* __restrict__ nf,
                                const float* __restrict__ bias,
                                float* __restrict__ out, int total) {
  int i = blockIdx.x * blockDim.x + threadIdx.x;
  if (i >= total) return;
  float v = out[i] + nf[i] + bias[i & (IN_FEAT - 1)];
  out[i] = v > 0.f ? v : 0.f;
}

// ===========================================================================
extern "C" void kernel_launch(void* const* d_in, const int* in_sizes, int n_in,
                              void* d_out, int out_size, void* d_ws, size_t ws_size,
                              hipStream_t stream) {
  (void)n_in; (void)out_size; (void)ws_size;
  const float* node_feat = (const float*)d_in[0];
  const float* edge_feat = (const float*)d_in[1];
  const int*   src       = (const int*)d_in[2];
  const int*   dst       = (const int*)d_in[3];
  const float* W         = (const float*)d_in[4];
  const float* We        = (const float*)d_in[5];
  const float* attn_l    = (const float*)d_in[6];
  const float* attn_r    = (const float*)d_in[7];
  const float* attn_e    = (const float*)d_in[8];
  const float* bias      = (const float*)d_in[9];
  float* out = (float*)d_out;

  const int N = in_sizes[0] / IN_FEAT;   // 50000
  const int E = in_sizes[2];             // 800000
  const int WEL = IN_FEAT * IN_FEAT;     // 65536 weight elements

  // workspace carve-up (~516 MB total)
  size_t off = 0;
  char* base = (char*)d_ws;
  auto carve = [&](size_t bytes) {
    char* p = base + off;
    off += (bytes + 255) & ~(size_t)255;
    return (void*)p;
  };
  float*    feat_src = (float*)carve((size_t)N * IN_FEAT * 4);
  float*    el       = (float*)carve((size_t)N * 8 * 4);
  float*    er       = (float*)carve((size_t)N * 8 * 4);
  __bf16*   feat_e   = (__bf16*)carve((size_t)E * IN_FEAT * 2);
  float*    ee       = (float*)carve((size_t)E * 8 * 4);
  float*    sc       = (float*)carve((size_t)E * 8 * 4);
  unsigned* smax     = (unsigned*)carve((size_t)N * 8 * 4);
  float*    denom    = (float*)carve((size_t)N * 8 * 4);
  __bf16*   wbf      = (__bf16*)carve((size_t)WEL * 2);
  __bf16*   webf     = (__bf16*)carve((size_t)WEL * 2);

  // allow >64KB dynamic LDS (CDNA5 WGP: up to 320KB per workgroup)
  (void)hipFuncSetAttribute((const void*)node_proj_kernel,
                            hipFuncAttributeMaxDynamicSharedMemorySize, SHMEM_BYTES);
  (void)hipFuncSetAttribute((const void*)edge_proj_kernel,
                            hipFuncAttributeMaxDynamicSharedMemorySize, SHMEM_BYTES);

  int nodeTiles = N / 16;                           // 3125 (exact)
  int edgeTiles = E / 16;                           // 50000 (exact)
  int nodeWG = (nodeTiles + 8 * NTPW - 1) / (8 * NTPW);   // 196
  int edgeWG = (edgeTiles + 8 * ETPW - 1) / (8 * ETPW);   // 1563

  cvt_weights_kernel<<<(WEL + 255) / 256, 256, 0, stream>>>(W, We, wbf, webf, WEL);
  init_kernel<<<(N * IN_FEAT + 255) / 256, 256, 0, stream>>>(
      out, smax, denom, N * IN_FEAT, N * 8);
  node_proj_kernel<<<nodeWG, 256, SHMEM_BYTES, stream>>>(
      node_feat, wbf, attn_l, attn_r, feat_src, el, er, nodeTiles);
  edge_proj_kernel<<<edgeWG, 256, SHMEM_BYTES, stream>>>(
      edge_feat, webf, attn_e, feat_e, ee, edgeTiles);
  score_kernel<<<(E * 8 + 255) / 256, 256, 0, stream>>>(
      src, dst, el, er, ee, sc, smax, E);
  expsum_kernel<<<(E * 8 + 255) / 256, 256, 0, stream>>>(
      dst, smax, sc, denom, E);
  message_kernel<<<E, 256, 0, stream>>>(
      src, dst, sc, denom, feat_src, feat_e, out, E);
  finalize_kernel<<<(N * IN_FEAT + 255) / 256, 256, 0, stream>>>(
      node_feat, bias, out, N * IN_FEAT);
}
